// DataEmbedding_13692355739751
// MI455X (gfx1250) — matrix operand order repaired and checked by reading, not compiled, not run on hardware
//
#include <hip/hip_runtime.h>
#include <hip/hip_bf16.h>

typedef __attribute__((ext_vector_type(16))) _Float16 v16h;
typedef __attribute__((ext_vector_type(8)))  float    v8f;
typedef __attribute__((ext_vector_type(4)))  unsigned int tdm_v4u;
typedef __attribute__((ext_vector_type(8)))  int          tdm_v8i;
typedef __attribute__((ext_vector_type(4)))  int          tdm_v4i;

#define B_   8
#define L_   2048
#define C_   7
#define D_   512
#define W_   24
#define NCH  56          // C_*(5+3)
#define LP   (L_ + 2)    // circular-padded length for conv
#define EPS_ 1e-5f

// ---------------------------------------------------------------------------
// Kernel 1: per-(b,l,c) feature extraction -> padded [B][56][L+2] buffer
// channels: 0-6 x, 7-13 mean, 14-20 max, 21-27 min, 28-34 std(ddof=1),
//           35-41 lag3, 42-48 lag5, 49-55 lag7
// ---------------------------------------------------------------------------
__global__ void feat_kernel(const float* __restrict__ x, float* __restrict__ feat) {
  int idx = blockIdx.x * blockDim.x + threadIdx.x;
  if (idx >= B_ * L_ * C_) return;
  int c = idx % C_;
  int l = (idx / C_) % L_;
  int b = idx / (C_ * L_);
  const float* xb = x + (size_t)b * L_ * C_ + c;   // stride C_ along l
  float xv = xb[(size_t)l * C_];

  float s = 0.f, s2 = 0.f, mn = 1e30f, mx = -1e30f;
  for (int w = 0; w < W_; ++w) {
    int li = l - (W_ - 1) + w;
    if (li < 0) li = 0;                            // edge-repeat padding
    float v = xb[(size_t)li * C_];
    s += v; s2 += v * v;
    mn = fminf(mn, v); mx = fmaxf(mx, v);
  }
  float mean = s / (float)W_;
  float var  = fmaxf((s2 - s * mean) / (float)(W_ - 1), 0.f);
  float sd   = sqrtf(var);
  float d3 = xv - xb[(size_t)(l >= 3 ? l - 3 : 0) * C_];
  float d5 = xv - xb[(size_t)(l >= 5 ? l - 5 : 0) * C_];
  float d7 = xv - xb[(size_t)(l >= 7 ? l - 7 : 0) * C_];

  float vals[8] = {xv, mean, mx, mn, sd, d3, d5, d7};
  for (int f = 0; f < 8; ++f) {
    float* row = feat + ((size_t)b * NCH + (c + 7 * f)) * LP;
    row[1 + l] = vals[f];
    if (l == 0)      row[L_ + 1] = vals[f];        // circular pad right
    if (l == L_ - 1) row[0]      = vals[f];        // circular pad left
  }
}

// ---------------------------------------------------------------------------
// Kernel 2: conv1d(k=3, 56->512) + bias + LayerNorm(gamma_c,beta_c)
// outputs: c_f32, c_f16, sq[b,l] = sum_d c^2
// one block per (b,l); 256 threads, 2 d's per thread
// ---------------------------------------------------------------------------
__global__ __launch_bounds__(256) void conv_ln_kernel(
    const float* __restrict__ feat, const float* __restrict__ conv_w,
    const float* __restrict__ conv_b, const float* __restrict__ gamma_c,
    const float* __restrict__ beta_c, float* __restrict__ cF32,
    _Float16* __restrict__ cF16, float* __restrict__ sqv) {
  int l = blockIdx.x % L_;
  int b = blockIdx.x / L_;
  int tid = threadIdx.x;
  __shared__ float sIn[NCH * 3];
  __shared__ float red[256];
  __shared__ float red2[256];

  if (tid < NCH * 3) {
    int ci = tid / 3, k = tid % 3;
    sIn[tid] = feat[((size_t)b * NCH + ci) * LP + l + k];
  }
  __syncthreads();

  float e[2];
  for (int j = 0; j < 2; ++j) {
    int d = tid + j * 256;
    const float* wrow = conv_w + (size_t)d * (NCH * 3);
    float acc = conv_b[d];
    #pragma unroll 4
    for (int i = 0; i < NCH * 3; ++i) acc += sIn[i] * wrow[i];
    e[j] = acc;
  }
  red[tid]  = e[0] + e[1];
  red2[tid] = e[0] * e[0] + e[1] * e[1];
  __syncthreads();
  for (int off = 128; off > 0; off >>= 1) {
    if (tid < off) { red[tid] += red[tid + off]; red2[tid] += red2[tid + off]; }
    __syncthreads();
  }
  float mu  = red[0] / (float)D_;
  float var = red2[0] / (float)D_ - mu * mu;
  float rs  = rsqrtf(fmaxf(var, 0.f) + EPS_);
  __syncthreads();

  float cv[2];
  for (int j = 0; j < 2; ++j) {
    int d = tid + j * 256;
    cv[j] = (e[j] - mu) * rs * gamma_c[d] + beta_c[d];
    size_t o = ((size_t)b * L_ + l) * D_ + d;
    cF32[o] = cv[j];
    cF16[o] = (_Float16)cv[j];
  }
  red[tid] = cv[0] * cv[0] + cv[1] * cv[1];
  __syncthreads();
  for (int off = 128; off > 0; off >>= 1) {
    if (tid < off) red[tid] += red[tid + off];
    __syncthreads();
  }
  if (tid == 0) sqv[(size_t)b * L_ + l] = red[0];
}

// ---------------------------------------------------------------------------
// Kernel 2b: tiled transpose c_f16 [B][L][D] -> cT [B][D][L]
// makes the S@C B-operand (lane = d, k = j) contiguous for b128 loads
// ---------------------------------------------------------------------------
__global__ __launch_bounds__(256) void transpose_kernel(
    const _Float16* __restrict__ cF16, _Float16* __restrict__ cT) {
  const int TL = L_ / 32;   // 64
  const int TD = D_ / 32;   // 16
  int t  = blockIdx.x;
  int b  = t / (TL * TD);
  int tt = t % (TL * TD);
  int tl = tt % TL;
  int td = tt / TL;
  int lx = threadIdx.x & 31;
  int ly = threadIdx.x >> 5;
  __shared__ _Float16 tile[32][33];
  const _Float16* src = cF16 + (size_t)b * L_ * D_;
  _Float16*       dst = cT   + (size_t)b * D_ * L_;
  #pragma unroll
  for (int r = ly; r < 32; r += 8)
    tile[r][lx] = src[(size_t)(tl * 32 + r) * D_ + td * 32 + lx];
  __syncthreads();
  #pragma unroll
  for (int r = ly; r < 32; r += 8)
    dst[(size_t)(td * 32 + r) * L_ + tl * 32 + lx] = tile[lx][r];
}

// ---------------------------------------------------------------------------
// Kernel 3: fused Gaussian-similarity attention (flash-style, WMMA f16->f32)
// sem[b,i,:] = sum_j exp(-max(dist,0)/2) * c[b,j,:] / rowsum
// one block per (b, 16-row block); 8 wave32
//  * A rows (16 x 512 f16) staged once into LDS by the Tensor Data Mover
//  * S@C B operand read contiguously from the transposed copy cT
//  * shared 16x128 fp16 S-tile consumed by all waves (64 D-cols each)
// ---------------------------------------------------------------------------
__global__ __launch_bounds__(256) void attn_kernel(
    const _Float16* __restrict__ cF16, const _Float16* __restrict__ cT,
    const float* __restrict__ sqv, float* __restrict__ sem) {
  const int nIB = L_ / 16;
  int b  = blockIdx.x / nIB;
  int i0 = (blockIdx.x % nIB) * 16;
  int tid  = threadIdx.x;
  int wave = tid >> 5;
  int lane = tid & 31;
  int m = lane & 15;       // A: M index / C,D: N index
  int h = lane >> 4;       // half-wave select
  const _Float16* cb  = cF16 + (size_t)b * L_ * D_;
  const _Float16* ctb = cT   + (size_t)b * D_ * L_;
  const float*    sqb = sqv  + (size_t)b * L_;

  __shared__ _Float16 sA[16][D_];    // A rows i0..i0+15, staged by TDM (16 KB)
  __shared__ _Float16 sS[16][128];   // fp16 S tile: rows i, cols 128 j's
  __shared__ float sRow[16];         // running row sums of S
  if (tid < 16) sRow[tid] = 0.f;

  // ---- TDM: DMA the 16x512 fp16 A-tile from global into LDS (wave 0 only)
  if (wave == 0) {
    unsigned ldsOff = (unsigned)(size_t)&sA[0][0];      // low 32 bits = LDS byte offset
    unsigned long long ga =
        (unsigned long long)(size_t)(cb + (size_t)i0 * D_);
    tdm_v4u g0 = { 1u,                                   // count=1, user D#
                   ldsOff,                               // lds_addr (bytes)
                   (unsigned)(ga & 0xffffffffull),       // global_addr[31:0]
                   (unsigned)((ga >> 32) & 0x1ffffffull) | (2u << 30) }; // [56:32] | type=2
    tdm_v8i g1 = { (int)(1u << 16),                      // wg_mask=0, data_size=2B
                   (int)((D_ & 0xffff) << 16),           // tensor_dim0[15:0]
                   (int)(((D_ >> 16) & 0xffff) | ((L_ & 0xffff) << 16)), // dim0 hi | dim1 lo
                   (int)(((L_ >> 16) & 0xffff) | (D_ << 16)),            // dim1 hi | tile_dim0=512
                   16,                                   // tile_dim1=16, tile_dim2=0
                   D_,                                   // tensor_dim0_stride[31:0]
                   0, 0 };                               // stride hi / dim1_stride
    tdm_v4i g2 = { 0, 0, 0, 0 };
    tdm_v4i g3 = { 0, 0, 0, 0 };
    tdm_v8i g4 = { 0, 0, 0, 0, 0, 0, 0, 0 };             // unused trailing group (clang-23 arity)
    __builtin_amdgcn_tensor_load_to_lds(g0, g1, g2, g3, g4, 0);
    __builtin_amdgcn_s_wait_tensorcnt(0);
  }
  __syncthreads();

  v8f acc[4];
  #pragma unroll
  for (int t = 0; t < 4; ++t)
    #pragma unroll
    for (int r = 0; r < 8; ++r) acc[t][r] = 0.f;
  int d0 = wave * 64;

  float sqi[8];
  #pragma unroll
  for (int r = 0; r < 8; ++r) sqi[r] = sqb[i0 + r + 8 * h];

  for (int jb = 0; jb < L_ / 128; ++jb) {
    int j0 = jb * 128 + wave * 16;

    // ---- Gram tile (16x16), K = D = 512 -> 16 WMMAs
    v8f g;
    #pragma unroll
    for (int r = 0; r < 8; ++r) g[r] = 0.f;
    for (int kb = 0; kb < D_; kb += 32) {
      v16h a, bm;
      const _Float16* arow = &sA[m][kb];
      const _Float16* brow = cb + (size_t)(j0 + m) * D_ + kb;  // C row j0+m: contiguous in k
      #pragma unroll
      for (int v = 0; v < 8; ++v) {
        int ko = ((v < 4) ? 0 : 16) + h * 8 + (v & 3) * 2;  // A f16 layout
        a[2 * v]     = arow[ko];
        a[2 * v + 1] = arow[ko + 1];
        // B layout: lane = N column, VGPR/pack = K (same order as A); B(k,n) = C[j0+n][kb+k]
        bm[2 * v]     = brow[ko];
        bm[2 * v + 1] = brow[ko + 1];
      }
      g = __builtin_amdgcn_wmma_f32_16x16x32_f16(false, a, false, bm,
                                                 (short)0, g, false, false);
    }

    // ---- S = exp(-max(dist,0)/2), write fp16 tile + rowsum accumulation
    float sv[8];
    float sqj = sqb[j0 + m];
    #pragma unroll
    for (int r = 0; r < 8; ++r) {
      float dist = sqi[r] + sqj - 2.f * g[r];
      sv[r] = __expf(-fmaxf(dist, 0.f) * 0.5f);
      sS[r + 8 * h][wave * 16 + m] = (_Float16)sv[r];
    }
    #pragma unroll
    for (int r = 0; r < 8; ++r) {
      float v = sv[r];
      v += __shfl_xor(v, 1, 32);
      v += __shfl_xor(v, 2, 32);
      v += __shfl_xor(v, 4, 32);
      v += __shfl_xor(v, 8, 32);       // 16-lane group now holds rowsum
      if (m == 0) atomicAdd(&sRow[r + 8 * h], v);
    }
    __syncthreads();

    // ---- acc += S(16x128) @ C[jb*128 .. +128)[d0 .. d0+64)
    int jrow0 = jb * 128;
    if (jb + 1 < L_ / 128)   // prefetch next j-block's C rows (global_prefetch_b8)
      __builtin_prefetch(cb + (size_t)(jrow0 + 128 + lane) * D_ + d0, 0, 1);
    for (int kk = 0; kk < 128; kk += 32) {
      v16h a;
      #pragma unroll
      for (int v = 0; v < 8; ++v) {
        int ko = kk + ((v < 4) ? 0 : 16) + h * 8 + (v & 3) * 2;
        a[2 * v]     = sS[m][ko];
        a[2 * v + 1] = sS[m][ko + 1];
      }
      const _Float16* brow = ctb + (size_t)(d0 + m) * L_ + jrow0 + kk;
      #pragma unroll
      for (int t = 0; t < 4; ++t) {
        v16h bm;
        #pragma unroll
        for (int v = 0; v < 8; ++v) {
          int kr = ((v < 4) ? 0 : 16) + h * 8 + (v & 3) * 2;
          bm[2 * v]     = brow[(size_t)t * 16 * L_ + kr];
          bm[2 * v + 1] = brow[(size_t)t * 16 * L_ + kr + 1];
        }
        acc[t] = __builtin_amdgcn_wmma_f32_16x16x32_f16(false, a, false, bm,
                                                        (short)0, acc[t],
                                                        false, false);
      }
    }
    __syncthreads();
  }

  // ---- epilogue: normalize by row sums, store sem (f32)
  #pragma unroll
  for (int t = 0; t < 4; ++t)
    #pragma unroll
    for (int r = 0; r < 8; ++r) {
      int M = r + 8 * h;
      int d = d0 + t * 16 + m;
      sem[((size_t)b * L_ + i0 + M) * D_ + d] = acc[t][r] / sRow[M];
    }
}

// ---------------------------------------------------------------------------
// Kernel 4: pe sinusoid, three LayerNorms, softmax-weighted blend
// one block per (b,l); 256 threads, 2 d's per thread
// ---------------------------------------------------------------------------
__device__ inline void mean_rstd(float v0, float v1, float* red, float* red2,
                                 int tid, float& mu, float& rs) {
  red[tid]  = v0 + v1;
  red2[tid] = v0 * v0 + v1 * v1;
  __syncthreads();
  for (int off = 128; off > 0; off >>= 1) {
    if (tid < off) { red[tid] += red[tid + off]; red2[tid] += red2[tid + off]; }
    __syncthreads();
  }
  float m   = red[0] / (float)D_;
  float var = red2[0] / (float)D_ - m * m;
  mu = m;
  rs = rsqrtf(fmaxf(var, 0.f) + EPS_);
  __syncthreads();
}

__global__ __launch_bounds__(256) void final_kernel(
    const float* __restrict__ cF32, const float* __restrict__ sem,
    const float* __restrict__ pe_learned, const float* __restrict__ wp,
    const float* __restrict__ gf, const float* __restrict__ bf,
    const float* __restrict__ gl, const float* __restrict__ bl,
    const float* __restrict__ gt, const float* __restrict__ bt,
    float* __restrict__ out) {
  int l = blockIdx.x % L_;
  int b = blockIdx.x / L_;
  int tid = threadIdx.x;
  __shared__ float red[256];
  __shared__ float red2[256];

  float pe[2], pel[2], tv[2], cv[2];
  const float negLogK = -logf(10000.f) / (float)D_;
  for (int j = 0; j < 2; ++j) {
    int d = tid + j * 256;
    int p = d >> 1;
    float ang = (float)l * __expf((float)(2 * p) * negLogK);
    pe[j]  = (d & 1) ? __cosf(ang) : __sinf(ang);
    pel[j] = pe_learned[(size_t)l * D_ + d];
    size_t o = ((size_t)b * L_ + l) * D_ + d;
    cv[j] = cF32[o];
    tv[j] = cv[j] + pe[j] + sem[o];
  }

  float muF, rsF, muL, rsL, muT, rsT;
  mean_rstd(pe[0],  pe[1],  red, red2, tid, muF, rsF);
  mean_rstd(pel[0], pel[1], red, red2, tid, muL, rsL);
  mean_rstd(tv[0],  tv[1],  red, red2, tid, muT, rsT);

  float w0 = wp[0], w1 = wp[1], w2 = wp[2], w3 = wp[3];
  float mx = fmaxf(fmaxf(w0, w1), fmaxf(w2, w3));
  float e0 = __expf(w0 - mx), e1 = __expf(w1 - mx);
  float e2 = __expf(w2 - mx), e3 = __expf(w3 - mx);
  float inv = 1.f / (e0 + e1 + e2 + e3);
  e0 *= inv; e1 *= inv; e2 *= inv; e3 *= inv;

  for (int j = 0; j < 2; ++j) {
    int d = tid + j * 256;
    size_t o = ((size_t)b * L_ + l) * D_ + d;
    float pef  = (pe[j]  - muF) * rsF * gf[d] + bf[d];
    float pelv = (pel[j] - muL) * rsL * gl[d] + bl[d];
    float tpe  = (tv[j]  - muT) * rsT * gt[d] + bt[d];
    out[o] = e0 * cv[j] + e1 * pef + e2 * pelv + e3 * tpe;
  }
}

// ---------------------------------------------------------------------------
extern "C" void kernel_launch(void* const* d_in, const int* in_sizes, int n_in,
                              void* d_out, int out_size, void* d_ws, size_t ws_size,
                              hipStream_t stream) {
  const float* x          = (const float*)d_in[0];
  const float* conv_w     = (const float*)d_in[1];
  const float* conv_b     = (const float*)d_in[2];
  const float* pe_learned = (const float*)d_in[3];
  const float* wp         = (const float*)d_in[4];
  const float* gamma_c    = (const float*)d_in[5];
  const float* beta_c     = (const float*)d_in[6];
  const float* gf         = (const float*)d_in[7];
  const float* bf         = (const float*)d_in[8];
  const float* gl         = (const float*)d_in[9];
  const float* bl         = (const float*)d_in[10];
  const float* gt         = (const float*)d_in[11];
  const float* bt         = (const float*)d_in[12];

  char* ws = (char*)d_ws;
  size_t off = 0;
  float* feat = (float*)(ws + off); off += (size_t)B_ * NCH * LP * sizeof(float);
  float* cF32 = (float*)(ws + off); off += (size_t)B_ * L_ * D_ * sizeof(float);
  float* sem  = (float*)(ws + off); off += (size_t)B_ * L_ * D_ * sizeof(float);
  float* sqv  = (float*)(ws + off); off += (size_t)B_ * L_ * sizeof(float);
  _Float16* cF16 = (_Float16*)(ws + off); off += (size_t)B_ * L_ * D_ * sizeof(_Float16);
  _Float16* cT   = (_Float16*)(ws + off); off += (size_t)B_ * L_ * D_ * sizeof(_Float16);

  int nfeat = B_ * L_ * C_;
  feat_kernel<<<(nfeat + 255) / 256, 256, 0, stream>>>(x, feat);
  conv_ln_kernel<<<B_ * L_, 256, 0, stream>>>(feat, conv_w, conv_b, gamma_c,
                                              beta_c, cF32, cF16, sqv);
  transpose_kernel<<<B_ * (L_ / 32) * (D_ / 32), 256, 0, stream>>>(cF16, cT);
  attn_kernel<<<B_ * (L_ / 16), 256, 0, stream>>>(cF16, cT, sqv, sem);
  final_kernel<<<B_ * L_, 256, 0, stream>>>(cF32, sem, pe_learned, wp, gf, bf,
                                            gl, bl, gt, bt, (float*)d_out);
}
